// GateCrossEncoder_29317446763256
// MI455X (gfx1250) — compile-verified, matrix operand order checked by reading, not run on hardware
//
#include <hip/hip_runtime.h>
#include <hip/hip_bf16.h>

#define NL    4
#define DM    512
#define NH    8
#define DEPTH 64
#define DFF   2048
#define BBATCH 16
#define LSEQ  512
#define MTOK  (BBATCH*LSEQ)   // 8192 tokens

typedef __attribute__((ext_vector_type(16))) __bf16 v16bf;
typedef __attribute__((ext_vector_type(8)))  float  v8f;
typedef unsigned short u16;
typedef unsigned int   u32;

// ---------- helpers ----------
__device__ __forceinline__ u16 f2bf(float f){
  u32 u = __float_as_uint(f);
  u += 0x7FFFu + ((u >> 16) & 1u);   // round-to-nearest-even
  return (u16)(u >> 16);
}
__device__ __forceinline__ float bf2f(u16 h){ return __uint_as_float(((u32)h) << 16); }

// bf16 fragment, reduction-contiguous source: lane covers K = kb..kb+7 and kb+16..kb+23
__device__ __forceinline__ v16bf load_frag_bf16(const u16* p){
  union { u32 w[8]; v16bf v; } f;
  const u32* a = (const u32*)p;
  const u32* b = (const u32*)(p + 16);
#pragma unroll
  for (int i = 0; i < 4; ++i){ f.w[i] = a[i]; f.w[4+i] = b[i]; }
  return f.v;
}
// build bf16 fragment from f32 data (used for LDS-resident softmax probs)
__device__ __forceinline__ v16bf frag_from_f32(const float* p0, const float* p1){
  union { u16 s[16]; v16bf v; } f;
#pragma unroll
  for (int i = 0; i < 8; ++i){ f.s[i] = f2bf(p0[i]); f.s[8+i] = f2bf(p1[i]); }
  return f.v;
}
__device__ __forceinline__ float wred32(float v){
#pragma unroll
  for (int m = 1; m < 32; m <<= 1) v += __shfl_xor(v, m, 32);
  return v;
}
__device__ __forceinline__ float wred16(float v){
#pragma unroll
  for (int m = 1; m < 16; m <<= 1) v += __shfl_xor(v, m, 32);
  return v;
}

// ---------- x/x2 + pos_enc ; H init ; x2 -> bf16 ----------
__global__ __launch_bounds__(256) void k_prep(const float* __restrict__ x, const float* __restrict__ x2,
                                              const float* __restrict__ pos, float* __restrict__ H,
                                              u16* __restrict__ x2bf){
  int idx = blockIdx.x*256 + threadIdx.x;
  if (idx >= MTOK*DM) return;
  int d = idx % DM;
  int t = idx / DM;
  int l = t % LSEQ;
  float pe = pos[l*DM + d];
  float xp = x[idx] + pe;
  H[(size_t)(t*2+0)*DM + d] = xp;
  H[(size_t)(t*2+1)*DM + d] = xp;
  x2bf[idx] = f2bf(x2[idx] + pe);
}

// ---------- fp32 W[K][N] -> bf16 Wt[N][K] ----------
__global__ __launch_bounds__(256) void k_wt(const float* __restrict__ W, u16* __restrict__ Wt, int K, int N){
  int idx = blockIdx.x*256 + threadIdx.x;
  if (idx >= K*N) return;
  int k = idx % K;
  int n = idx / K;
  Wt[idx] = f2bf(W[(size_t)k*N + n]);
}

// ---------- hyper-pre: H_avg, coeff projections, h0 = sum Am*H, LayerNorm -> bf16 ----------
__global__ __launch_bounds__(256) void k_hyper(const float* __restrict__ H,
    const float* AmS, const float* ArS, const float* BS, const float* alp,
    const float* Wam, const float* bam, const float* War, const float* bar,
    const float* Wb, const float* bbv,
    const float* g, const float* be,
    u16* __restrict__ lnout, float* __restrict__ ArO, float* __restrict__ BcO){
  int t = blockIdx.x*8 + (threadIdx.x >> 5);   // one wave32 per token
  int lane = threadIdx.x & 31;
  float A0[16], A1[16];
  float am0=0, am1=0, ar0=0, ar1=0, ar2=0, ar3=0, bb0=0, bb1=0;
#pragma unroll
  for (int j = 0; j < 16; ++j){
    int d = lane + 32*j;
    float h0v = H[(size_t)(t*2+0)*DM + d];
    float h1v = H[(size_t)(t*2+1)*DM + d];
    A0[j] = h0v; A1[j] = h1v;
    float hv = 0.5f*(h0v + h1v);
    am0 += hv*Wam[d*2+0]; am1 += hv*Wam[d*2+1];
    ar0 += hv*War[d*4+0]; ar1 += hv*War[d*4+1];
    ar2 += hv*War[d*4+2]; ar3 += hv*War[d*4+3];
    bb0 += hv*Wb[d*2+0];  bb1 += hv*Wb[d*2+1];
  }
  am0 = wred32(am0); am1 = wred32(am1);
  ar0 = wred32(ar0); ar1 = wred32(ar1); ar2 = wred32(ar2); ar3 = wred32(ar3);
  bb0 = wred32(bb0); bb1 = wred32(bb1);
  float a = alp[0];
  float s0 = AmS[0] + tanhf(am0 + bam[0])*a;
  float s1 = AmS[1] + tanhf(am1 + bam[1])*a;
  float mx = fmaxf(s0, s1);
  float e0 = __expf(s0-mx), e1 = __expf(s1-mx);
  float Am0 = e0/(e0+e1), Am1 = 1.0f - Am0;
  float h0[16]; float sum=0.f, sq=0.f;
#pragma unroll
  for (int j=0;j<16;++j){ float v = Am0*A0[j] + Am1*A1[j]; h0[j]=v; sum+=v; sq+=v*v; }
  sum = wred32(sum); sq = wred32(sq);
  float mu = sum*(1.0f/DM);
  float var = sq*(1.0f/DM) - mu*mu;
  float rstd = rsqrtf(var + 1e-6f);
#pragma unroll
  for (int j=0;j<16;++j){
    int d = lane + 32*j;
    lnout[(size_t)t*DM + d] = f2bf((h0[j]-mu)*rstd*g[d] + be[d]);
  }
  if (lane == 0){
    ArO[t*4+0] = ArS[0] + tanhf(ar0 + bar[0])*a;
    ArO[t*4+1] = ArS[1] + tanhf(ar1 + bar[1])*a;
    ArO[t*4+2] = ArS[2] + tanhf(ar2 + bar[2])*a;
    ArO[t*4+3] = ArS[3] + tanhf(ar3 + bar[3])*a;
    BcO[t*2+0] = BS[0] + tanhf(bb0 + bbv[0])*a;
    BcO[t*2+1] = BS[1] + tanhf(bb1 + bbv[1])*a;
  }
}

// ---------- generic bf16 WMMA GEMM: C[MxN] = A[MxK] * Wt[NxK]^T + bias ----------
// B (weight) tile is double-buffered in LDS via async-to-LDS DMA (ASYNCcnt path):
// each thread owns one of the 128 rows; 4x b128 async ops move its 64B k-slice.
#define GF_RELU  1
#define GF_OBF16 2

__global__ __launch_bounds__(128) void k_gemm(const u16* __restrict__ A, int lda,
                                              const u16* __restrict__ Bt, int ldb,
                                              const float* __restrict__ bias,
                                              void* __restrict__ Cv, int ldc,
                                              int K, int flags){
  __shared__ u16 Btile[2][128*32];   // 2 x 8KB double buffer
  int tid  = threadIdx.x;
  int wave = tid >> 5;
  int lane = tid & 31;
  int r = lane & 15, half = lane >> 4;
  int mW = blockIdx.x*128 + (wave & 1)*64;
  int nBaseBlk = blockIdx.y*128;
  int wn = (wave >> 1)*64;           // wave's column offset within the block tile

  // async issue of one 128x32 bf16 B-tile into LDS buffer `buf` at reduction offset k
  auto issueB = [&](int buf, int k){
    unsigned la = (unsigned)(size_t)(&Btile[buf][tid*32]);
    unsigned long long ga = (unsigned long long)(Bt + (size_t)(nBaseBlk + tid)*ldb + k);
    // IOFFSET applies to both the LDS and global side (ISA 15.18.3)
    asm volatile("global_load_async_to_lds_b128 %0, %1, off"           :: "v"(la), "v"(ga) : "memory");
    asm volatile("global_load_async_to_lds_b128 %0, %1, off offset:16" :: "v"(la), "v"(ga) : "memory");
    asm volatile("global_load_async_to_lds_b128 %0, %1, off offset:32" :: "v"(la), "v"(ga) : "memory");
    asm volatile("global_load_async_to_lds_b128 %0, %1, off offset:48" :: "v"(la), "v"(ga) : "memory");
  };

  v8f acc[4][4] = {};
  issueB(0, 0);
  for (int k = 0; k < K; k += 32){
    int buf = (k >> 5) & 1;
    asm volatile("s_wait_asynccnt 0x0" ::: "memory");   // own rows landed
    __syncthreads();                                    // whole tile visible to all waves
    if (k + 32 < K) issueB(buf ^ 1, k + 32);            // overlap next DMA with compute
    if (k + 64 < K)
      __builtin_prefetch(A + (size_t)(mW + r)*lda + k + 64, 0, 1);  // global_prefetch_b8

    int kb = k + half*8;
    v16bf af[4];
#pragma unroll
    for (int mt = 0; mt < 4; ++mt)
      af[mt] = load_frag_bf16(A + (size_t)(mW + mt*16 + r)*lda + kb);
#pragma unroll
    for (int nt = 0; nt < 4; ++nt){
      const u16* bp = &Btile[buf][(wn + nt*16 + r)*32 + half*8];   // ds_load_b128 x2
      v16bf bfrag = load_frag_bf16(bp);
#pragma unroll
      for (int mt = 0; mt < 4; ++mt)
        acc[mt][nt] = __builtin_amdgcn_wmma_f32_16x16x32_bf16(
            false, af[mt], false, bfrag, (short)0, acc[mt][nt], false, false);
    }
  }
  bool relu = (flags & GF_RELU) != 0, obf = (flags & GF_OBF16) != 0;
#pragma unroll
  for (int mt = 0; mt < 4; ++mt){
#pragma unroll
    for (int nt = 0; nt < 4; ++nt){
      int col = nBaseBlk + wn + nt*16 + r;
      float bv = bias ? bias[col] : 0.0f;
#pragma unroll
      for (int rr = 0; rr < 8; ++rr){
        int row = mW + mt*16 + half*8 + rr;
        float v = acc[mt][nt][rr] + bv;
        if (relu) v = fmaxf(v, 0.0f);
        if (obf) ((u16*)Cv)[(size_t)row*ldc + col] = f2bf(v);
        else     ((float*)Cv)[(size_t)row*ldc + col] = v;
      }
    }
  }
}

// ---------- (Q+V) transposed per head: QpVt[b,h,d,l] ----------
__global__ __launch_bounds__(256) void k_qpvt(const u16* __restrict__ Q, const u16* __restrict__ V,
                                              u16* __restrict__ O){
  int idx = blockIdx.x*256 + threadIdx.x;
  if (idx >= BBATCH*NH*DEPTH*LSEQ) return;
  int l = idx & (LSEQ-1);
  int d = (idx >> 9) & (DEPTH-1);
  int h = (idx >> 15) & (NH-1);
  int b = idx >> 18;
  size_t src = (size_t)(b*LSEQ + l)*DM + h*DEPTH + d;
  O[idx] = f2bf(bf2f(Q[src]) + bf2f(V[src]));
}

// ---------- fused attention: scores->LDS, softmax in LDS, P@(Q+V)^T, sigmoid gate ----------
__global__ __launch_bounds__(256) void k_attn(const u16* __restrict__ Q, const u16* __restrict__ Kb,
                                              const u16* __restrict__ QpVt,
                                              const float* __restrict__ wg, const float* __restrict__ bg,
                                              u16* __restrict__ O){
  __shared__ float S[64*512];            // 128 KB of the 320 KB WGP LDS
  int blk = blockIdx.x;
  int qb = blk & 7;
  int h  = (blk >> 3) & 7;
  int b  = blk >> 6;
  int wave = threadIdx.x >> 5, lane = threadIdx.x & 31;
  int r = lane & 15, half = lane >> 4;
  int q0 = qb*64;

  { // Phase 1: S = (Q_tile @ K^T) / sqrt(64); 8 waves, each owns 64 cols of keys
    int colBase = wave*64;
    v16bf af[4][2];
#pragma unroll
    for (int mt = 0; mt < 4; ++mt){
      const u16* qp = Q + (size_t)(b*LSEQ + q0 + mt*16 + r)*DM + h*DEPTH + half*8;
      af[mt][0] = load_frag_bf16(qp);
      af[mt][1] = load_frag_bf16(qp + 32);
    }
#pragma unroll
    for (int nt = 0; nt < 4; ++nt){
      int key = colBase + nt*16 + r;
      const u16* kp = Kb + (size_t)(b*LSEQ + key)*DM + h*DEPTH + half*8;
      v16bf bf0 = load_frag_bf16(kp);
      v16bf bf1 = load_frag_bf16(kp + 32);
#pragma unroll
      for (int mt = 0; mt < 4; ++mt){
        v8f acc = {};
        acc = __builtin_amdgcn_wmma_f32_16x16x32_bf16(false, af[mt][0], false, bf0, (short)0, acc, false, false);
        acc = __builtin_amdgcn_wmma_f32_16x16x32_bf16(false, af[mt][1], false, bf1, (short)0, acc, false, false);
#pragma unroll
        for (int rr = 0; rr < 8; ++rr)
          S[(mt*16 + half*8 + rr)*512 + colBase + nt*16 + r] = acc[rr]*0.125f;
      }
    }
  }
  __syncthreads();
  // Phase 2: row softmax in LDS (f32, in place)
  if (threadIdx.x < 64){
    float* row = S + threadIdx.x*512;
    float mx = -3.4e38f;
    for (int i = 0; i < 512; ++i) mx = fmaxf(mx, row[i]);
    float sum = 0.f;
    for (int i = 0; i < 512; ++i){ float e = __expf(row[i]-mx); row[i] = e; sum += e; }
    float inv = 1.0f/sum;
    for (int i = 0; i < 512; ++i) row[i] *= inv;
  }
  __syncthreads();
  // Phase 3: out = P @ (Q+V)^T ; gate ; store. 4 waves, 16 q-rows each, full 64 depth cols.
  if (wave < 4){
    v8f acc[4] = {};
    const float* rowp = S + (wave*16 + r)*512;
    for (int ks = 0; ks < 16; ++ks){
      int kb = ks*32 + half*8;
      v16bf afr = frag_from_f32(rowp + kb, rowp + kb + 16);
#pragma unroll
      for (int nt = 0; nt < 4; ++nt){
        const u16* bp = QpVt + (size_t)((b*NH + h)*DEPTH + nt*16 + r)*LSEQ + kb;
        v16bf bfr = load_frag_bf16(bp);
        acc[nt] = __builtin_amdgcn_wmma_f32_16x16x32_bf16(false, afr, false, bfr, (short)0, acc[nt], false, false);
      }
    }
    float bgv = bg[0];
#pragma unroll
    for (int rr = 0; rr < 8; ++rr){
      float p = 0.f;
#pragma unroll
      for (int nt = 0; nt < 4; ++nt) p += acc[nt][rr]*wg[nt*16 + r];
      p = wred16(p);                                    // 16-lane reduce (per row group)
      float gv = 1.0f/(1.0f + __expf(-(p + bgv)));
      int lr = wave*16 + half*8 + rr;
      size_t rowo = (size_t)(b*LSEQ + q0 + lr)*DM + h*DEPTH;
#pragma unroll
      for (int nt = 0; nt < 4; ++nt)
        O[rowo + nt*16 + r] = f2bf(acc[nt][rr]*gv);
    }
  }
}

// ---------- hyper combine: Hout[i] = Bc[i]*y + sum_j Ar[j][i]*H[j] ----------
__global__ __launch_bounds__(256) void k_combine(const float* __restrict__ Hin, const float* __restrict__ y,
                                                 const float* __restrict__ Ar, const float* __restrict__ Bc,
                                                 float* __restrict__ Hout){
  int idx = blockIdx.x*256 + threadIdx.x;
  if (idx >= MTOK*DM) return;
  int d = idx % DM, t = idx / DM;
  float h0v = Hin[(size_t)(t*2+0)*DM + d];
  float h1v = Hin[(size_t)(t*2+1)*DM + d];
  float yv = y[idx];
  float a00 = Ar[t*4+0], a01 = Ar[t*4+1], a10 = Ar[t*4+2], a11 = Ar[t*4+3];
  Hout[(size_t)(t*2+0)*DM + d] = Bc[t*2+0]*yv + a00*h0v + a10*h1v;
  Hout[(size_t)(t*2+1)*DM + d] = Bc[t*2+1]*yv + a01*h0v + a11*h1v;
}

__global__ __launch_bounds__(256) void k_final(const float* __restrict__ H, float* __restrict__ out){
  int idx = blockIdx.x*256 + threadIdx.x;
  if (idx >= MTOK*DM) return;
  int d = idx % DM, t = idx / DM;
  out[idx] = H[(size_t)(t*2+0)*DM + d] + H[(size_t)(t*2+1)*DM + d];
}

// ---------- host ----------
extern "C" void kernel_launch(void* const* d_in, const int* in_sizes, int n_in,
                              void* d_out, int out_size, void* d_ws, size_t ws_size,
                              hipStream_t stream){
  (void)in_sizes; (void)n_in; (void)out_size; (void)ws_size;
  const float* x   = (const float*)d_in[0];
  const float* x2  = (const float*)d_in[1];
  const float* pos = (const float*)d_in[2 + NL*40];   // pos_enc after 4 layers x 40 params

  char* wsp = (char*)d_ws;
  size_t off = 0;
  auto alloc = [&](size_t bytes)->void*{
    void* p = wsp + off;
    off += (bytes + 255) & ~(size_t)255;
    return p;
  };
  float* Hc   = (float*)alloc((size_t)MTOK*2*DM*4);
  float* Hn   = (float*)alloc((size_t)MTOK*2*DM*4);
  u16* x2bf   = (u16*)alloc((size_t)MTOK*DM*2);
  u16* lnbf   = (u16*)alloc((size_t)MTOK*DM*2);
  float* ArB  = (float*)alloc((size_t)MTOK*4*4);
  float* BcB  = (float*)alloc((size_t)MTOK*2*4);
  u16* Qbf    = (u16*)alloc((size_t)MTOK*DM*2);
  u16* Kbf    = (u16*)alloc((size_t)MTOK*DM*2);
  u16* Vbf    = (u16*)alloc((size_t)MTOK*DM*2);
  u16* QpVt   = (u16*)alloc((size_t)MTOK*DM*2);
  u16* Atbf   = (u16*)alloc((size_t)MTOK*DM*2);
  u16* Tmpbf  = (u16*)alloc((size_t)MTOK*DM*2);
  float* Ybuf = (float*)alloc((size_t)MTOK*DM*4);
  u16* Hidbf  = (u16*)alloc((size_t)MTOK*DFF*2);
  u16* wqt = (u16*)alloc((size_t)DM*DM*2);
  u16* wkt = (u16*)alloc((size_t)DM*DM*2);
  u16* wvt = (u16*)alloc((size_t)DM*DM*2);
  u16* wot = (u16*)alloc((size_t)DM*DM*2);
  u16* wct = (u16*)alloc((size_t)DM*DM*2);
  u16* w1t = (u16*)alloc((size_t)DM*DFF*2);
  u16* w2t = (u16*)alloc((size_t)DFF*DM*2);

  const int nEl = MTOK*DM;
  k_prep<<<nEl/256, 256, 0, stream>>>(x, x2, pos, Hc, x2bf);

  for (int layer = 0; layer < NL; ++layer){
    const float* P[40];
    for (int i = 0; i < 40; ++i) P[i] = (const float*)d_in[2 + layer*40 + i];
    // 0 wq 1 bq 2 wk 3 bk 4 wv 5 bv 6 wo 7 bo 8 wg 9 bg 10 w_cond 11 b_cond
    // 12 w1 13 b1 14 w2 15 b2 16 ln1_g 17 ln1_b 18 ln2_g 19 ln2_b
    // 20..29 hc_atten {AmS,ArS,BS,alpha,Wam,bam,War,bar,Wb,bb} ; 30..39 hc_ffn

    k_wt<<<(DM*DM)/256,256,0,stream>>>(P[0],  wqt, DM, DM);
    k_wt<<<(DM*DM)/256,256,0,stream>>>(P[2],  wkt, DM, DM);
    k_wt<<<(DM*DM)/256,256,0,stream>>>(P[4],  wvt, DM, DM);
    k_wt<<<(DM*DM)/256,256,0,stream>>>(P[6],  wot, DM, DM);
    k_wt<<<(DM*DM)/256,256,0,stream>>>(P[10], wct, DM, DM);
    k_wt<<<(DM*DFF)/256,256,0,stream>>>(P[12], w1t, DM, DFF);
    k_wt<<<(DFF*DM)/256,256,0,stream>>>(P[14], w2t, DFF, DM);

    // attention branch
    k_hyper<<<MTOK/8, 256, 0, stream>>>(Hc, P[20],P[21],P[22],P[23],P[24],P[25],P[26],P[27],P[28],P[29],
                                        P[16], P[17], lnbf, ArB, BcB);
    dim3 g512(MTOK/128, DM/128);
    k_gemm<<<g512,128,0,stream>>>(x2bf, DM, wqt, DM, P[1], Qbf, DM, DM, GF_OBF16);
    k_gemm<<<g512,128,0,stream>>>(lnbf, DM, wkt, DM, P[3], Kbf, DM, DM, GF_OBF16);
    k_gemm<<<g512,128,0,stream>>>(lnbf, DM, wvt, DM, P[5], Vbf, DM, DM, GF_OBF16);
    k_qpvt<<<nEl/256,256,0,stream>>>(Qbf, Vbf, QpVt);
    k_attn<<<BBATCH*NH*(LSEQ/64), 256, 0, stream>>>(Qbf, Kbf, QpVt, P[8], P[9], Atbf);
    k_gemm<<<g512,128,0,stream>>>(Atbf,  DM, wot, DM, P[7],  Tmpbf, DM, DM, GF_OBF16);
    k_gemm<<<g512,128,0,stream>>>(Tmpbf, DM, wct, DM, P[11], Ybuf,  DM, DM, 0);
    k_combine<<<nEl/256,256,0,stream>>>(Hc, Ybuf, ArB, BcB, Hn);

    // FFN branch
    k_hyper<<<MTOK/8, 256, 0, stream>>>(Hn, P[30],P[31],P[32],P[33],P[34],P[35],P[36],P[37],P[38],P[39],
                                        P[18], P[19], lnbf, ArB, BcB);
    dim3 gFF(MTOK/128, DFF/128);
    k_gemm<<<gFF,128,0,stream>>>(lnbf,  DM,  w1t, DM,  P[13], Hidbf, DFF, DM,  GF_OBF16|GF_RELU);
    k_gemm<<<g512,128,0,stream>>>(Hidbf, DFF, w2t, DFF, P[15], Ybuf,  DM,  DFF, 0);
    k_combine<<<nEl/256,256,0,stream>>>(Hn, Ybuf, ArB, BcB, Hc);
  }
  k_final<<<nEl/256,256,0,stream>>>(Hc, (float*)d_out);
}
